// EnhancedMSTSN_60146722013282
// MI455X (gfx1250) — matrix operand order repaired and checked by reading, not compiled, not run on hardware
//
#include <hip/hip_runtime.h>
#include <hip/hip_bf16.h>
#include <math.h>

// ---------------- constants ----------------
#define BB 4
#define TT 24
#define NN 400
#define BT 96          // B*T
#define BTN 38400      // B*T*N
#define BN 1600        // B*N

typedef __attribute__((ext_vector_type(16))) _Float16 v16h;
typedef __attribute__((ext_vector_type(8)))  float    v8f;
typedef __attribute__((ext_vector_type(4)))  float    v4f;

// ---------------- weight pre-pack into WMMA B-fragment layout ----------------
// out[((tn*KT + c)*32 + lane)*16 + e] = f16(W[(c*32 + (e<8 ? half*8 + e : 16 + half*8 + e-8))*NC + tn*16 + l16])
// so each lane's 16-half B fragment is contiguous (32 bytes -> vector load in GEMM).
__global__ void pack_w_kernel(const float* __restrict__ W, _Float16* __restrict__ out,
                              int K, int NC) {
  int KT = (K + 31) / 32;
  int total = (NC / 16) * KT * 32 * 16;
  int idx = blockIdx.x * blockDim.x + threadIdx.x;
  if (idx >= total) return;
  int e    = idx & 15;
  int lane = (idx >> 4) & 31;
  int rest = idx >> 9;
  int c  = rest % KT;
  int tn = rest / KT;
  int half = lane >> 4, l16 = lane & 15;
  int kb = c * 32 + ((e < 8) ? (half * 8 + e) : (16 + half * 8 + (e - 8)));
  int n  = tn * 16 + l16;
  out[idx] = (kb < K) ? (_Float16)W[(size_t)kb * NC + n] : (_Float16)0.f;
}

// ---------------- WMMA GEMM: C[M,NC] = epi(A[M,K] @ W[K,NC] + bias) + resid ----------------
// one wave per 16-row strip (all NC columns); K, NC, bias/residual presence compile-time;
// A loaded as b128 vectors, converted f32->f16 in-register; B fragments pre-packed f16.
// EPI: 0 = none, 2 = exact GELU (after bias, before residual).
template<int EPI, int K, int NC, bool HB, bool HR>
__global__ __launch_bounds__(32)
void gemm_wmma(const float* __restrict__ A, const _Float16* __restrict__ Wp,
               const float* __restrict__ bias, const float* __restrict__ resid,
               float* __restrict__ C, int M) {
  constexpr int KT = (K + 31) / 32;
  const int tm = blockIdx.x;
  const int lane = threadIdx.x;
  const int half = lane >> 4, l16 = lane & 15;
  const float* Arow = A + (size_t)(tm * 16 + l16) * K;

  // A fragments for all K-chunks (16x32 layout): two contiguous 8-float runs per chunk.
  v16h af[KT];
#pragma unroll
  for (int c = 0; c < KT; ++c) {
    const float* p0 = Arow + c * 32 + half * 8;        // run0: always valid (c*32+16 <= K)
    v4f a0 = *(const v4f*)(p0);
    v4f a1 = *(const v4f*)(p0 + 4);
    v4f a2 = {0.f, 0.f, 0.f, 0.f};
    v4f a3 = {0.f, 0.f, 0.f, 0.f};
    if (c * 32 + 32 <= K) {                            // run1 validity: compile-time
      const float* p1 = Arow + c * 32 + 16 + half * 8;
      a2 = *(const v4f*)(p1);
      a3 = *(const v4f*)(p1 + 4);
    }
#pragma unroll
    for (int t = 0; t < 4; ++t) {
      af[c][t]      = (_Float16)a0[t];
      af[c][4 + t]  = (_Float16)a1[t];
      af[c][8 + t]  = (_Float16)a2[t];
      af[c][12 + t] = (_Float16)a3[t];
    }
  }

  const v16h* Bfrag = (const v16h*)Wp;
#pragma unroll
  for (int tn = 0; tn < NC / 16; ++tn) {
    v8f acc = {};
#pragma unroll
    for (int c = 0; c < KT; ++c) {
      v16h bf = Bfrag[(tn * KT + c) * 32 + lane];      // 32B contiguous per lane
      acc = __builtin_amdgcn_wmma_f32_16x16x32_f16(false, af[c], false, bf,
                                                   (short)0, acc, false, false);
    }
    const int n = tn * 16 + l16;
    const float bval = HB ? bias[n] : 0.f;             // hoisted: one load per (lane, tn)
    const int mbase = tm * 16 + half * 8;
    float rv[8];
    if (HR) {                                          // straight-line residual loads (clause)
#pragma unroll
      for (int v = 0; v < 8; ++v) rv[v] = resid[(size_t)(mbase + v) * NC + n];
    }
#pragma unroll
    for (int v = 0; v < 8; ++v) {
      float val = acc[v];
      if (HB) val += bval;
      if (EPI == 2) val = 0.5f * val * (1.f + erff(val * 0.70710678118654752f));
      if (HR) val += rv[v];
      C[(size_t)(mbase + v) * NC + n] = val;
    }
  }
}

// ---------------- adjacency ----------------
__global__ void adj_norm_kernel(const float* __restrict__ e, float* __restrict__ nrm) {
  int i = blockIdx.x * blockDim.x + threadIdx.x;
  if (i >= NN) return;
  float s = 0.f;
#pragma unroll
  for (int c = 0; c < 16; ++c) { float v = e[i * 16 + c]; s += v * v; }
  nrm[i] = sqrtf(s) + 1e-12f;
}

__global__ void adj_mask_kernel(const float* __restrict__ e, const float* __restrict__ nrm,
                                unsigned char* __restrict__ mask) {
  int idx = blockIdx.x * blockDim.x + threadIdx.x;
  if (idx >= NN * NN) return;
  int j = idx % NN, i = idx / NN;
  float d = 0.f;
#pragma unroll
  for (int c = 0; c < 16; ++c) d += e[i * 16 + c] * e[j * 16 + c];
  mask[idx] = (d > 0.5f * nrm[i] * nrm[j]) ? 1 : 0;
}

// ---------------- input projection: h = x @ proj_w + proj_b ----------------
__global__ void proj_kernel(const float* __restrict__ x, const float* __restrict__ w,
                            const float* __restrict__ b, float* __restrict__ h) {
  int idx = blockIdx.x * blockDim.x + threadIdx.x;
  if (idx >= BTN * 16) return;
  int c = idx & 15, r = idx >> 4;
  float s = b[c];
#pragma unroll
  for (int f = 0; f < 3; ++f) s += x[r * 3 + f] * w[f * 16 + c];
  h[idx] = s;
}

// ---------------- GAT source/target terms ----------------
__global__ void gat_esed_kernel(const float* __restrict__ hw, const float* __restrict__ asrc,
                                const float* __restrict__ adst, float* __restrict__ es,
                                float* __restrict__ ed, int H, int D) {
  int idx = blockIdx.x * blockDim.x + threadIdx.x;   // (bt*H + h)*N + n
  if (idx >= BT * 4 * NN) return;
  int n = idx % NN; int t = idx / NN; int h = t % H; int bt = t / H;
  const float* hwp = hw + ((size_t)(bt * NN + n) * H + h) * D;
  float s = 0.f, d2 = 0.f;
  for (int d = 0; d < D; ++d) { float v = hwp[d]; s += v * asrc[h * D + d]; d2 += v * adst[h * D + d]; }
  es[idx] = s; ed[idx] = d2;
}

// ---------------- fused masked-GAT aggregation (online softmax), wave per (bt,i) ----------------
template<int D>
__global__ __launch_bounds__(32)
void gat_agg_kernel(const float* __restrict__ hw, const float* __restrict__ es,
                    const float* __restrict__ ed, const unsigned char* __restrict__ mask,
                    const float* __restrict__ bias, float* __restrict__ out,
                    int H, int doRelu) {
  int row = blockIdx.x;               // bt*N + i
  int bt = row / NN, i = row - bt * NN;
  int lane = threadIdx.x;
  const unsigned char* mrow = mask + (size_t)i * NN;
  for (int h = 0; h < H; ++h) {
    float edi = ed[((size_t)bt * H + h) * NN + i];
    const float* esrow = es + ((size_t)bt * H + h) * NN;
    float m = -1e30f, s = 0.f;
    float acc[D];
#pragma unroll
    for (int d = 0; d < D; ++d) acc[d] = 0.f;
    for (int j = lane; j < NN; j += 32) {
      if (!mrow[j]) continue;
      float x = edi + esrow[j];
      x = (x > 0.f) ? x : 0.2f * x;          // leaky_relu(0.2)
      float mn = fmaxf(m, x);
      float sc = __expf(m - mn);
      float e  = __expf(x - mn);
      s = s * sc + e;
      const float* hwp = hw + ((size_t)(bt * NN + j) * H + h) * D;
#pragma unroll
      for (int d = 0; d < D; ++d) acc[d] = acc[d] * sc + e * hwp[d];
      m = mn;
    }
    // merge online-softmax state across the wave
#pragma unroll
    for (int off = 16; off; off >>= 1) {
      float m2 = __shfl_xor(m, off, 32);
      float s2 = __shfl_xor(s, off, 32);
      float mn = fmaxf(m, m2);
      float sc1 = __expf(m - mn), sc2 = __expf(m2 - mn);
      s = s * sc1 + s2 * sc2;
#pragma unroll
      for (int d = 0; d < D; ++d) {
        float a2 = __shfl_xor(acc[d], off, 32);
        acc[d] = acc[d] * sc1 + a2 * sc2;
      }
      m = mn;
    }
    if (lane == 0) {
      float inv = 1.f / s;
#pragma unroll
      for (int d = 0; d < D; ++d) {
        float v = acc[d] * inv + bias[h * D + d];
        if (doRelu) v = fmaxf(v, 0.f);
        out[((size_t)row * H + h) * D + d] = v;
      }
    }
  }
}

// ---------------- LayerNorm over C=32, wave per row ----------------
__global__ void layernorm32_kernel(const float* __restrict__ x, const float* __restrict__ g,
                                   const float* __restrict__ b, float* __restrict__ y, int rows) {
  int row = blockIdx.x * blockDim.y + threadIdx.y;
  if (row >= rows) return;
  int lane = threadIdx.x;
  float v = x[(size_t)row * 32 + lane];
  float s = v;
#pragma unroll
  for (int off = 16; off; off >>= 1) s += __shfl_xor(s, off, 32);
  float mean = s * 0.03125f;
  float d = v - mean;
  float q = d * d;
#pragma unroll
  for (int off = 16; off; off >>= 1) q += __shfl_xor(q, off, 32);
  float var = q * 0.03125f;
  y[(size_t)row * 32 + lane] = d * rsqrtf(var + 1e-3f) * g[lane] + b[lane];
}

// ---------------- temporal attention, T=24, 2 heads x key_dim 16, block per (b,n) ----------------
__global__ void temporal_attn_kernel(const float* __restrict__ q, const float* __restrict__ k,
                                     const float* __restrict__ v, float* __restrict__ o) {
  __shared__ float ks[2][TT][16];
  __shared__ float vs[2][TT][16];
  int bn = blockIdx.x;
  int tid = threadIdx.x;
  size_t base = (size_t)bn * TT * 32;
  for (int idx = tid; idx < TT * 32; idx += 64) {
    int t = idx >> 5, c = idx & 31;
    int h = c >> 4, kk = c & 15;
    ks[h][t][kk] = k[base + t * 32 + c];
    vs[h][t][kk] = v[base + t * 32 + c];
  }
  __syncthreads();
  int h = tid >> 5;
  int lane = tid & 31;
  if (lane < TT) {
    int t = lane;
    float qr[16];
#pragma unroll
    for (int kk = 0; kk < 16; ++kk) qr[kk] = q[base + t * 32 + h * 16 + kk];
    float sc[TT]; float mx = -1e30f;
    for (int s2 = 0; s2 < TT; ++s2) {
      float d = 0.f;
#pragma unroll
      for (int kk = 0; kk < 16; ++kk) d += qr[kk] * ks[h][s2][kk];
      d *= 0.25f;                       // 1/sqrt(16)
      sc[s2] = d;
      mx = fmaxf(mx, d);
    }
    float ssum = 0.f;
    for (int s2 = 0; s2 < TT; ++s2) { sc[s2] = __expf(sc[s2] - mx); ssum += sc[s2]; }
    float inv = 1.f / ssum;
    for (int kk = 0; kk < 16; ++kk) {
      float a = 0.f;
      for (int s2 = 0; s2 < TT; ++s2) a += sc[s2] * vs[h][s2][kk];
      o[base + t * 32 + h * 16 + kk] = a * inv;
    }
  }
}

// ---------------- means over T ----------------
__global__ void mean_spatial_kernel(const float* __restrict__ sp, float* __restrict__ sf) {
  int idx = blockIdx.x * blockDim.x + threadIdx.x;   // (b*N+n)*32 + c
  if (idx >= BN * 32) return;
  int c = idx & 31; int t = idx >> 5; int n = t % NN; int b = t / NN;
  float s = 0.f;
  for (int tt = 0; tt < TT; ++tt) s += sp[(((size_t)(b * TT + tt)) * NN + n) * 32 + c];
  sf[idx] = s * (1.f / TT);
}
__global__ void mean_temporal_kernel(const float* __restrict__ tp, float* __restrict__ tf) {
  int idx = blockIdx.x * blockDim.x + threadIdx.x;
  if (idx >= BN * 32) return;
  int c = idx & 31; int t = idx >> 5; int n = t % NN; int b = t / NN;
  const float* p = tp + ((size_t)(b * NN + n) * TT) * 32 + c;
  float s = 0.f;
  for (int tt = 0; tt < TT; ++tt) s += p[tt * 32];
  tf[idx] = s * (1.f / TT);
}

// ---------------- cross attention: wave per (b,i,h), 2 heads x key_dim 32 ----------------
__global__ __launch_bounds__(32)
void cross_attn_kernel(const float* __restrict__ q, const float* __restrict__ k,
                       const float* __restrict__ v, float* __restrict__ o) {
  __shared__ float sacc[32][33];
  int gid = blockIdx.x;
  int h = gid & 1; int t = gid >> 1; int i = t % NN; int b = t / NN;
  int lane = threadIdx.x;
  size_t qbase = ((size_t)(b * NN + i)) * 64 + h * 32;
  float qr[32];
#pragma unroll
  for (int c = 0; c < 32; ++c) qr[c] = q[qbase + c];
  float m = -1e30f, s = 0.f;
  float acc[32];
#pragma unroll
  for (int c = 0; c < 32; ++c) acc[c] = 0.f;
  for (int j = lane; j < NN; j += 32) {
    size_t kb = ((size_t)(b * NN + j)) * 64 + h * 32;
    float dot = 0.f;
#pragma unroll
    for (int c = 0; c < 32; ++c) dot += qr[c] * k[kb + c];
    float x = dot * 0.17677669529663687f;   // 1/sqrt(32)
    float mn = fmaxf(m, x);
    float sc = __expf(m - mn);
    float e  = __expf(x - mn);
    s = s * sc + e;
#pragma unroll
    for (int c = 0; c < 32; ++c) acc[c] = acc[c] * sc + e * v[kb + c];
    m = mn;
  }
  float mg = m;
#pragma unroll
  for (int off = 16; off; off >>= 1) mg = fmaxf(mg, __shfl_xor(mg, off, 32));
  float sc = __expf(m - mg);
  s *= sc;
#pragma unroll
  for (int off = 16; off; off >>= 1) s += __shfl_xor(s, off, 32);
#pragma unroll
  for (int c = 0; c < 32; ++c) sacc[lane][c] = acc[c] * sc;
  __syncthreads();
  float r = 0.f;
#pragma unroll
  for (int l = 0; l < 32; ++l) r += sacc[l][lane];
  o[((size_t)(b * NN + i)) * 64 + h * 32 + lane] = r / s;
}

// ---------------- final regression: out = r1 @ reg2_w + reg2_b ----------------
__global__ void reg2_kernel(const float* __restrict__ r1, const float* __restrict__ w,
                            const float* __restrict__ b, float* __restrict__ out) {
  int r = blockIdx.x * blockDim.x + threadIdx.x;
  if (r >= BN) return;
  float s = b[0];
#pragma unroll
  for (int c = 0; c < 16; ++c) s += r1[r * 16 + c] * w[c];
  out[r] = s;
}

// ---------------- launch ----------------
extern "C" void kernel_launch(void* const* d_in, const int* in_sizes, int n_in,
                              void* d_out, int out_size, void* d_ws, size_t ws_size,
                              hipStream_t stream) {
  (void)in_sizes; (void)n_in; (void)out_size; (void)ws_size;
  const float* x         = (const float*)d_in[0];
  const float* adj_embed = (const float*)d_in[1];
  const float* proj_w    = (const float*)d_in[2];
  const float* proj_b    = (const float*)d_in[3];
  const float* gat1_w    = (const float*)d_in[4];
  const float* gat1_asrc = (const float*)d_in[5];
  const float* gat1_adst = (const float*)d_in[6];
  const float* gat1_b    = (const float*)d_in[7];
  const float* gat2_w    = (const float*)d_in[8];
  const float* gat2_asrc = (const float*)d_in[9];
  const float* gat2_adst = (const float*)d_in[10];
  const float* gat2_b    = (const float*)d_in[11];
  const float* t_ln1_g   = (const float*)d_in[12];
  const float* t_ln1_b   = (const float*)d_in[13];
  const float* t_wq      = (const float*)d_in[14];
  const float* t_bq      = (const float*)d_in[15];
  const float* t_wk      = (const float*)d_in[16];
  const float* t_bk      = (const float*)d_in[17];
  const float* t_wv      = (const float*)d_in[18];
  const float* t_bv      = (const float*)d_in[19];
  const float* t_wo      = (const float*)d_in[20];
  const float* t_bo      = (const float*)d_in[21];
  const float* t_ln2_g   = (const float*)d_in[22];
  const float* t_ln2_b   = (const float*)d_in[23];
  const float* t_ff1_w   = (const float*)d_in[24];
  const float* t_ff1_b   = (const float*)d_in[25];
  const float* t_ff2_w   = (const float*)d_in[26];
  const float* t_ff2_b   = (const float*)d_in[27];
  const float* t_lnf_g   = (const float*)d_in[28];
  const float* t_lnf_b   = (const float*)d_in[29];
  const float* ca_wq     = (const float*)d_in[30];
  const float* ca_bq     = (const float*)d_in[31];
  const float* ca_wk     = (const float*)d_in[32];
  const float* ca_bk     = (const float*)d_in[33];
  const float* ca_wv     = (const float*)d_in[34];
  const float* ca_bv     = (const float*)d_in[35];
  const float* ca_wo     = (const float*)d_in[36];
  const float* ca_bo     = (const float*)d_in[37];
  const float* reg1_w    = (const float*)d_in[38];
  const float* reg1_b    = (const float*)d_in[39];
  const float* reg2_w    = (const float*)d_in[40];
  const float* reg2_b    = (const float*)d_in[41];
  float* out = (float*)d_out;

  // workspace layout (floats; all chunks multiples of 256B)
  float* ws = (float*)d_ws;
  size_t off = 0;
  auto take   = [&](size_t nf) { float* p = ws + off; off += (nf + 63) & ~(size_t)63; return p; };
  auto take16 = [&](size_t nh) { _Float16* p = (_Float16*)(ws + off); off += (nh / 2 + 63) & ~(size_t)63; return p; };
  float*         nrm   = take(NN);
  unsigned char* mask  = (unsigned char*)take(NN * NN / 4);   // 160000 bytes
  float* hbuf  = take((size_t)BTN * 16);   // h, then reused for g1
  float* hw1   = take((size_t)BTN * 16);
  float* es1   = take((size_t)BT * 4 * NN);
  float* ed1   = take((size_t)BT * 4 * NN);
  float* hw2   = take((size_t)BTN * 32);
  float* spat  = take((size_t)BTN * 32);   // spatial [B,T,N,32] flat (== tin reshape)
  float* c3    = take((size_t)BTN * 32);   // hn -> attn_o -> h2n -> (h1+ff)
  float* c4    = take((size_t)BTN * 32);   // q -> h1 -> tout
  float* c5    = take((size_t)BTN * 32);   // k -> ffmid (lower half)
  float* c6    = take((size_t)BTN * 32);   // v -> ffmid (upper half, contiguous with c5)
  float* sfeat = take((size_t)BN * 32);
  float* tfeat = take((size_t)BN * 32);
  float* caq   = take((size_t)BN * 64);
  float* cak   = take((size_t)BN * 64);
  float* cav   = take((size_t)BN * 64);
  float* cao   = take((size_t)BN * 64);
  float* fused = take((size_t)BN * 32);
  float* r1    = take((size_t)BN * 16);
  // packed f16 weights (WMMA B-fragment layout): (NC/16)*ceil(K/32)*512 halves each
  _Float16* pk_gat1 = take16(512);    // 16x16
  _Float16* pk_gat2 = take16(1024);   // 16x32
  _Float16* pk_twq  = take16(1024);   // 32x32
  _Float16* pk_twk  = take16(1024);
  _Float16* pk_twv  = take16(1024);
  _Float16* pk_two  = take16(1024);
  _Float16* pk_ff1  = take16(2048);   // 32x64
  _Float16* pk_ff2  = take16(2048);   // 64x32
  _Float16* pk_caq  = take16(2048);   // 32x64
  _Float16* pk_cak  = take16(2048);
  _Float16* pk_cav  = take16(2048);
  _Float16* pk_cawo = take16(2048);   // 64x32
  _Float16* pk_reg1 = take16(512);    // 32x16

  auto pack = [&](const float* W, _Float16* dst, int K, int NC) {
    int KT = (K + 31) / 32;
    int total = (NC / 16) * KT * 32 * 16;
    pack_w_kernel<<<(total + 255) / 256, 256, 0, stream>>>(W, dst, K, NC);
  };
  pack(gat1_w, pk_gat1, 16, 16);
  pack(gat2_w, pk_gat2, 16, 32);
  pack(t_wq,   pk_twq,  32, 32);
  pack(t_wk,   pk_twk,  32, 32);
  pack(t_wv,   pk_twv,  32, 32);
  pack(t_wo,   pk_two,  32, 32);
  pack(t_ff1_w, pk_ff1, 32, 64);
  pack(t_ff2_w, pk_ff2, 64, 32);
  pack(ca_wq,  pk_caq,  32, 64);
  pack(ca_wk,  pk_cak,  32, 64);
  pack(ca_wv,  pk_cav,  32, 64);
  pack(ca_wo,  pk_cawo, 64, 32);
  pack(reg1_w, pk_reg1, 32, 16);

  // 1) adjacency mask
  adj_norm_kernel<<<(NN + 63) / 64, 64, 0, stream>>>(adj_embed, nrm);
  adj_mask_kernel<<<(NN * NN + 255) / 256, 256, 0, stream>>>(adj_embed, nrm, mask);

  // 2) input projection -> h [BTN,16]
  proj_kernel<<<(BTN * 16 + 255) / 256, 256, 0, stream>>>(x, proj_w, proj_b, hbuf);

  // 3) GAT layer 1 (H=4, D=4) with ReLU
  gemm_wmma<0, 16, 16, false, false><<<BTN / 16, 32, 0, stream>>>(hbuf, pk_gat1, nullptr, nullptr, hw1, BTN);
  gat_esed_kernel<<<(BT * 4 * NN + 255) / 256, 256, 0, stream>>>(hw1, gat1_asrc, gat1_adst, es1, ed1, 4, 4);
  gat_agg_kernel<4><<<BTN, 32, 0, stream>>>(hw1, es1, ed1, mask, gat1_b, hbuf, 4, 1);   // g1 -> hbuf

  // 4) GAT layer 2 (H=4, D=8) -> spatial [BTN,32]
  gemm_wmma<0, 16, 32, false, false><<<BTN / 16, 32, 0, stream>>>(hbuf, pk_gat2, nullptr, nullptr, hw2, BTN);
  gat_esed_kernel<<<(BT * 4 * NN + 255) / 256, 256, 0, stream>>>(hw2, gat2_asrc, gat2_adst, es1, ed1, 4, 8);
  gat_agg_kernel<8><<<BTN, 32, 0, stream>>>(hw2, es1, ed1, mask, gat2_b, spat, 4, 0);

  // 5) temporal transformer (reshape to [B*N, T, 32] is a no-op on the flat buffer)
  layernorm32_kernel<<<BTN / 8, dim3(32, 8), 0, stream>>>(spat, t_ln1_g, t_ln1_b, c3, BTN);          // hn
  gemm_wmma<0, 32, 32, true, false><<<BTN / 16, 32, 0, stream>>>(c3, pk_twq, t_bq, nullptr, c4, BTN);   // q
  gemm_wmma<0, 32, 32, true, false><<<BTN / 16, 32, 0, stream>>>(c3, pk_twk, t_bk, nullptr, c5, BTN);   // k
  gemm_wmma<0, 32, 32, true, false><<<BTN / 16, 32, 0, stream>>>(c3, pk_twv, t_bv, nullptr, c6, BTN);   // v
  temporal_attn_kernel<<<BN, 64, 0, stream>>>(c4, c5, c6, c3);                                          // attn o -> c3
  gemm_wmma<0, 32, 32, true, true><<<BTN / 16, 32, 0, stream>>>(c3, pk_two, t_bo, spat, c4, BTN);       // h1 = tin + o@wo
  layernorm32_kernel<<<BTN / 8, dim3(32, 8), 0, stream>>>(c4, t_ln2_g, t_ln2_b, c3, BTN);               // h2n
  gemm_wmma<2, 32, 64, true, false><<<BTN / 16, 32, 0, stream>>>(c3, pk_ff1, t_ff1_b, nullptr, c5, BTN); // gelu(ff1) -> c5..c6
  gemm_wmma<0, 64, 32, true, true><<<BTN / 16, 32, 0, stream>>>(c5, pk_ff2, t_ff2_b, c4, c3, BTN);       // h1 + ff
  layernorm32_kernel<<<BTN / 8, dim3(32, 8), 0, stream>>>(c3, t_lnf_g, t_lnf_b, c4, BTN);                // tout

  // 6) pooled features
  mean_spatial_kernel<<<(BN * 32 + 255) / 256, 256, 0, stream>>>(spat, sfeat);
  mean_temporal_kernel<<<(BN * 32 + 255) / 256, 256, 0, stream>>>(c4, tfeat);

  // 7) cross attention (H=2, key_dim=32)
  gemm_wmma<0, 32, 64, true, false><<<BN / 16, 32, 0, stream>>>(sfeat, pk_caq, ca_bq, nullptr, caq, BN);
  gemm_wmma<0, 32, 64, true, false><<<BN / 16, 32, 0, stream>>>(tfeat, pk_cak, ca_bk, nullptr, cak, BN);
  gemm_wmma<0, 32, 64, true, false><<<BN / 16, 32, 0, stream>>>(tfeat, pk_cav, ca_bv, nullptr, cav, BN);
  cross_attn_kernel<<<BB * NN * 2, 32, 0, stream>>>(caq, cak, cav, cao);
  gemm_wmma<0, 64, 32, true, false><<<BN / 16, 32, 0, stream>>>(cao, pk_cawo, ca_bo, nullptr, fused, BN);

  // 8) regression head
  gemm_wmma<2, 32, 16, true, false><<<BN / 16, 32, 0, stream>>>(fused, pk_reg1, reg1_b, nullptr, r1, BN);
  reg2_kernel<<<(BN + 255) / 256, 256, 0, stream>>>(r1, reg2_w, reg2_b, out);
}
